// WindowAttention_21947282882647
// MI455X (gfx1250) — matrix operand order repaired and checked by reading, not compiled, not run on hardware
//
#include <hip/hip_runtime.h>
#include <hip/hip_bf16.h>

// ---- problem constants ----
#define CDIM   256
#define HDIM   224
#define WDIM   224
#define BDIM   4
#define WIN    8
#define NWX    28            // 224/8
#define NWIN_PER_IMG (NWX*NWX)   // 784
#define NWIN_TOTAL  (BDIM*NWIN_PER_IMG) // 3136
#define TOK    64            // tokens per window
#define HEADS  8
#define DHEAD  32
#define WQKV_N (768*256)     // 196608
#define WOUT_N (256*256)     // 65536
#define XSTR   68            // padded LDS row stride (floats) for [c][t] buffers

typedef _Float16 v16h __attribute__((ext_vector_type(16)));
typedef _Float16 v8h  __attribute__((ext_vector_type(8)));
typedef _Float16 v4h  __attribute__((ext_vector_type(4)));
typedef float    v8f  __attribute__((ext_vector_type(8)));
typedef unsigned int v4u __attribute__((ext_vector_type(4)));
typedef int          v8i __attribute__((ext_vector_type(8)));
typedef int          v4i __attribute__((ext_vector_type(4)));

// LDS layout (dynamic, 262144 bytes total):
//   [0      , 32768)  bufX   : f16 [64][256]  xn, later merged attention output o
//   [32768  , 98304)  bufQK  : f16 [64][512]  q (cols 0..255), k (cols 256..511)
//   [98304  , 131072) vT     : f16 [256][64]  v transposed (d-major)
//   [131072 , 262144) bufS   : f32 scratch:
//        - TDM-loaded raw x as [c][68] (69632 B), then
//        - per-wave 16KB score slabs (8 x 64x64 f32), then
//        - final f32 out transposed [256][68]
#define OFF_X    0
#define OFF_QK   32768
#define OFF_VT   98304
#define OFF_S    131072
#define SMEM_BYTES 262144

__device__ __forceinline__ v8f vzero8() {
    v8f z;
#pragma unroll
    for (int i = 0; i < 8; ++i) z[i] = 0.f;
    return z;
}

__device__ __forceinline__ v8f wmma_f16(v16h a, v16h b, v8f c) {
    // D = A(16x32 f16) * B(32x16 f16) + C(16x16 f32)
    return __builtin_amdgcn_wmma_f32_16x16x32_f16(
        /*neg_a=*/false, a, /*neg_b=*/false, b,
        /*c_mod=*/(short)0, c, /*reuse_a=*/false, /*reuse_b=*/false);
}

// A-operand (16-bit, 16x32): lane<16 holds row M=lane, halves 0..15 = K base+0..15,
// where base = (lane>=16 ? 16 : 0).
__device__ __forceinline__ v16h load_a_frag(const _Float16* p) {
    v8h lo = *(const v8h*)(p);
    v8h hi = *(const v8h*)(p + 8);
    v16h r;
#pragma unroll
    for (int i = 0; i < 8; ++i) { r[i] = lo[i]; r[i + 8] = hi[i]; }
    return r;
}

// ---------------------------------------------------------------------------
// Kernel 1: convert f32 weights -> f16 workspace (wqkv then wout).
// ---------------------------------------------------------------------------
__global__ void cvt_weights_kernel(const float* __restrict__ wqkv,
                                   const float* __restrict__ wout,
                                   _Float16* __restrict__ dst) {
    int idx = blockIdx.x * 256 + threadIdx.x;
    if (idx < WQKV_N) {
        dst[idx] = (_Float16)wqkv[idx];
    } else if (idx < WQKV_N + WOUT_N) {
        dst[idx] = (_Float16)wout[idx - WQKV_N];
    }
}

// ---------------------------------------------------------------------------
// Kernel 2: one workgroup (256 threads, 8 wave32) per 8x8 window.
// ---------------------------------------------------------------------------
__global__ __launch_bounds__(256)
void win_attn_kernel(const float* __restrict__ x,
                     const float* __restrict__ ln_gamma,
                     const float* __restrict__ ln_beta,
                     const float* __restrict__ b_qkv,
                     const float* __restrict__ b_out,
                     const _Float16* __restrict__ wqkv_h,   // [768][256]
                     const _Float16* __restrict__ wout_h,   // [256][256]
                     float* __restrict__ out) {
    extern __shared__ char smem[];
    _Float16* bufX  = (_Float16*)(smem + OFF_X);
    _Float16* bufQK = (_Float16*)(smem + OFF_QK);
    _Float16* vT    = (_Float16*)(smem + OFF_VT);
    float*    bufS  = (float*)  (smem + OFF_S);
    float*    xrawT = bufS;                   // f32 [256][XSTR], token-contiguous

    const int tid  = threadIdx.x;
    // wave id is uniform across the wave: pin it to an SGPR so wave-derived
    // control flow (q/k vs v stores) becomes scalar branches, and wave-indexed
    // LDS addressing is scalarized.
    const int wave = __builtin_amdgcn_readfirstlane(tid >> 5);
    const int lane = tid & 31;
    const int lane16 = lane & 15;
    const int hiHalf = (lane >= 16);          // 0 or 1

    // ---- window decode ----
    const int wid = blockIdx.x;
    const int b   = wid / NWIN_PER_IMG;
    const int rem = wid % NWIN_PER_IMG;
    const int h0  = (rem / NWX) * WIN;
    const int w0  = (rem % NWX) * WIN;

    // ---------------- phase 1: DMA the 8x8x256 patch into LDS -------------
    // 3-D tile: x = 8 contiguous floats, y = 8 rows (stride 224),
    // z = 256 channels (stride 224*224). TDM iteration (x,y,z) lands it in
    // LDS as [c][t]; pad 4 DWORDs every 64 DWORDs -> row stride 68 floats.
#if __has_builtin(__builtin_amdgcn_tensor_load_to_lds) && __has_builtin(__builtin_amdgcn_s_wait_tensorcnt)
    if (wave == 0) {
        const unsigned lds_addr = (unsigned)(unsigned long long)(uintptr_t)xrawT;
        const unsigned long long ga =
            (unsigned long long)(uintptr_t)(x + (((size_t)b * CDIM) * HDIM + h0) * WDIM + w0);
        v4u g0;
        g0[0] = 1u;                                   // count=1, user mode
        g0[1] = lds_addr;                             // lds_addr[31:0]
        g0[2] = (unsigned)ga;                         // global_addr[31:0]
        g0[3] = (unsigned)(ga >> 32) | (2u << 30);    // global_addr[56:32] | type=2
        v8i g1;
        g1[0] = (int)((2u << 16)        // data_size = 4 bytes
                    | (1u << 20)        // pad_enable
                    | (5u << 22)        // pad_interval: 64 DWORDs
                    | (3u << 25));      // pad_amount: 4 DWORDs
        g1[1] = (int)(8u << 16);        // tensor_dim0 = 8 (low 16 bits)
        g1[2] = (int)(8u << 16);        // tensor_dim1 = 8 (low 16 bits)
        g1[3] = (int)(8u << 16);        // tile_dim0 = 8
        g1[4] = (int)(8u | (256u << 16)); // tile_dim1 = 8, tile_dim2 = 256
        g1[5] = (int)224u;              // tensor_dim0_stride = W
        g1[6] = (int)(50176u << 16);    // tensor_dim1_stride = H*W (low 16 bits)
        g1[7] = 0;                      // tensor_dim1_stride high bits
        v4i g2;
        g2[0] = 256;                    // tensor_dim2
        g2[1] = 0; g2[2] = 0; g2[3] = 0;
        v4i g3; g3[0] = 0; g3[1] = 0; g3[2] = 0; g3[3] = 0;
        v8i g4;                          // extra arg in 6-arg builtin variant
#pragma unroll
        for (int i = 0; i < 8; ++i) g4[i] = 0;
        __builtin_amdgcn_tensor_load_to_lds(g0, g1, g2, g3, g4, 0);
        __builtin_amdgcn_s_wait_tensorcnt(0);
    }
#else
    // Fallback: manual coalesced load, thread = channel.
    {
        const int c = tid;
        const float* base = x + (((size_t)b * CDIM + c) * HDIM + h0) * WDIM + w0;
        float* drow = xrawT + c * XSTR;
#pragma unroll
        for (int r = 0; r < WIN; ++r) {
            const float4* s4 = (const float4*)(base + (size_t)r * WDIM);
            float4 a0 = s4[0], a1 = s4[1];
            *(float4*)(drow + r * 8)     = a0;
            *(float4*)(drow + r * 8 + 4) = a1;
        }
    }
#endif
    __syncthreads();

    // ---------------- phase 2: LayerNorm -> f16 xn in bufX ----------------
    {
        const int t = tid >> 2;               // token 0..63
        const int p = tid & 3;                // quarter of channels
        const float* xc = xrawT + (p * 64) * XSTR + t;   // c = p*64 + i
        float sum = 0.f, ss = 0.f;
#pragma unroll
        for (int i = 0; i < 64; ++i) {
            float v = xc[i * XSTR];
            sum += v;
            ss  += v * v;
        }
#pragma unroll
        for (int off = 1; off < 4; off <<= 1) {
            sum += __shfl_xor(sum, off, 32);
            ss  += __shfl_xor(ss,  off, 32);
        }
        const float mean = sum * (1.f / 256.f);
        const float var  = ss * (1.f / 256.f) - mean * mean;
        const float rstd = rsqrtf(var + 1e-5f);
        _Float16* xo = bufX + t * CDIM + p * 64;
        const float4* g4  = (const float4*)(ln_gamma + p * 64);
        const float4* bb4 = (const float4*)(ln_beta + p * 64);
#pragma unroll
        for (int i = 0; i < 16; ++i) {
            float4 g = g4[i];
            float4 bt = bb4[i];
            float vx = xc[(4 * i + 0) * XSTR];
            float vy = xc[(4 * i + 1) * XSTR];
            float vz = xc[(4 * i + 2) * XSTR];
            float vw = xc[(4 * i + 3) * XSTR];
            v4h h;
            h[0] = (_Float16)((vx - mean) * rstd * g.x + bt.x);
            h[1] = (_Float16)((vy - mean) * rstd * g.y + bt.y);
            h[2] = (_Float16)((vz - mean) * rstd * g.z + bt.z);
            h[3] = (_Float16)((vw - mean) * rstd * g.w + bt.w);
            *(v4h*)(xo + 4 * i) = h;
        }
    }
    __syncthreads();

    // ---------------- phase 3: QKV GEMM (64x256 @ 256x768) ----------------
    // 48 n-tiles of 16 cols; wave handles 6. q,k -> bufQK, v -> vT (transposed).
    {
        const int kb_b = hiHalf ? 16 : 0;     // B operand K base per lane
        const int kb_a = hiHalf ? 16 : 0;     // A operand K base per lane
        for (int i = 0; i < 6; ++i) {
            const int nt = wave * 6 + i;                // wave-uniform (SGPR)
            const int coln = nt * 16 + lane16;          // out channel 0..767
            const _Float16* wrow = wqkv_h + coln * 256; // weight row (K-contig)
            v16h bf[8];
#pragma unroll
            for (int k = 0; k < 8; ++k)
                bf[k] = *(const v16h*)(wrow + k * 32 + kb_b);
            const float bias = b_qkv[coln];
            for (int m = 0; m < 4; ++m) {
                const int rowq = m * 16 + lane16;
                const _Float16* ap = bufX + rowq * CDIM + kb_a;
                v16h af[8];                    // batch A loads: 16 b128 in flight
#pragma unroll
                for (int k = 0; k < 8; ++k)
                    af[k] = load_a_frag(ap + k * 32);
                v8f acc = vzero8();
#pragma unroll
                for (int k = 0; k < 8; ++k)
                    acc = wmma_f16(af[k], bf[k], acc);
                const int r0 = m * 16 + (hiHalf ? 8 : 0);
                if (nt < 32) {                 // q or k (scalar branch: nt uniform)
#pragma unroll
                    for (int e = 0; e < 8; ++e)
                        bufQK[(r0 + e) * 512 + coln] = (_Float16)(acc[e] + bias);
                } else {                       // v -> transposed [d][token]
                    const int cv = coln - 512;
                    v8h hv;
#pragma unroll
                    for (int e = 0; e < 8; ++e)
                        hv[e] = (_Float16)(acc[e] + bias);
                    *(v8h*)(vT + cv * TOK + r0) = hv;   // one b128 store
                }
            }
        }
    }
    __syncthreads();

    // ---------------- phase 4: per-head attention (wave = head) -----------
    const int hd = wave;
    float* waveS = bufS + wave * (TOK * TOK); // 64x64 f32 slab
    {
        const int kb_a = hiHalf ? 16 : 0;
        // scores = q @ k^T (K = d = 32, single WMMA per 16x16 tile)
        for (int mq = 0; mq < 4; ++mq) {
            const int rowq = mq * 16 + lane16;
            v16h af = load_a_frag(bufQK + rowq * 512 + hd * DHEAD + kb_a);
#pragma unroll
            for (int nk = 0; nk < 4; ++nk) {
                const int keyc = nk * 16 + lane16;
                v16h bk = *(const v16h*)(bufQK + keyc * 512 + 256 + hd * DHEAD +
                                         (hiHalf ? 16 : 0));
                v8f acc = vzero8();
                acc = wmma_f16(af, bk, acc);
                const int r0 = mq * 16 + (hiHalf ? 8 : 0);
                const int col = nk * 16 + lane16;
#pragma unroll
                for (int e = 0; e < 8; ++e)
                    waveS[(r0 + e) * TOK + col] = acc[e];
            }
        }
        // softmax: each lane owns 2 rows; in-place f32 exp then f16 attn
        // written at row-stride 128 halves (aliases only the lane's own row).
        const float scale = 0.17677669529663687f;  // 1/sqrt(32)
        _Float16* attnH = (_Float16*)waveS;
#pragma unroll
        for (int rr = 0; rr < 2; ++rr) {
            const int r = lane * 2 + rr;
            float* rp = waveS + r * TOK;
            float mx = -3.0e38f;
#pragma unroll
            for (int i = 0; i < 16; ++i) {
                float4 v = ((const float4*)rp)[i];
                mx = fmaxf(mx, fmaxf(fmaxf(v.x, v.y), fmaxf(v.z, v.w)));
            }
            float sum = 0.f;
#pragma unroll
            for (int i = 0; i < 16; ++i) {
                float4 v = ((const float4*)rp)[i];
                v.x = __expf(scale * (v.x - mx));
                v.y = __expf(scale * (v.y - mx));
                v.z = __expf(scale * (v.z - mx));
                v.w = __expf(scale * (v.w - mx));
                ((float4*)rp)[i] = v;
                sum += v.x + v.y + v.z + v.w;
            }
            const float inv = 1.f / sum;
            _Float16* dst = attnH + r * 128;
#pragma unroll
            for (int i = 0; i < 16; ++i) {
                float4 v = ((const float4*)rp)[i];
                v4h h;
                h[0] = (_Float16)(v.x * inv);
                h[1] = (_Float16)(v.y * inv);
                h[2] = (_Float16)(v.z * inv);
                h[3] = (_Float16)(v.w * inv);
                *(v4h*)(dst + 4 * i) = h;
            }
        }
        // o = attn(64x64) @ v(64x32); vT gives K-contiguous B operand.
        const int kb_a2 = hiHalf ? 16 : 0;
        for (int mq = 0; mq < 4; ++mq) {
#pragma unroll
            for (int nt = 0; nt < 2; ++nt) {
                v8f acc = vzero8();
                const int rowq = mq * 16 + lane16;
                const int cd = hd * DHEAD + nt * 16 + lane16;
#pragma unroll
                for (int ks = 0; ks < 2; ++ks) {
                    v16h af = load_a_frag(attnH + rowq * 128 + ks * 32 + kb_a2);
                    v16h bv = *(const v16h*)(vT + cd * TOK + ks * 32 +
                                             (hiHalf ? 16 : 0));
                    acc = wmma_f16(af, bv, acc);
                }
                const int r0 = mq * 16 + (hiHalf ? 8 : 0);
#pragma unroll
                for (int e = 0; e < 8; ++e)
                    bufX[(r0 + e) * CDIM + cd] = (_Float16)acc[e];
            }
        }
    }
    __syncthreads();

    // ---------------- phase 5: output projection (64x256 @ 256x256) -------
    // Result staged transposed: outFT [256][XSTR] f32, per-lane contiguous.
    {
        float* outFT = bufS;
        const int kb_b = hiHalf ? 16 : 0;
        const int kb_a = hiHalf ? 16 : 0;
#pragma unroll
        for (int i = 0; i < 2; ++i) {
            const int nt = wave * 2 + i;
            const int coln = nt * 16 + lane16;
            const _Float16* wrow = wout_h + coln * 256;
            v16h bf[8];
#pragma unroll
            for (int k = 0; k < 8; ++k)
                bf[k] = *(const v16h*)(wrow + k * 32 + kb_b);
            const float bias = b_out[coln];
            for (int m = 0; m < 4; ++m) {
                const int rowq = m * 16 + lane16;
                const _Float16* ap = bufX + rowq * CDIM + kb_a;
                v16h af[8];                    // batch A loads
#pragma unroll
                for (int k = 0; k < 8; ++k)
                    af[k] = load_a_frag(ap + k * 32);
                v8f acc = vzero8();
#pragma unroll
                for (int k = 0; k < 8; ++k)
                    acc = wmma_f16(af[k], bf[k], acc);
                const int r0 = m * 16 + (hiHalf ? 8 : 0);
                float* dst = outFT + coln * XSTR + r0;
                *(float4*)(dst)     = make_float4(acc[0] + bias, acc[1] + bias,
                                                  acc[2] + bias, acc[3] + bias);
                *(float4*)(dst + 4) = make_float4(acc[4] + bias, acc[5] + bias,
                                                  acc[6] + bias, acc[7] + bias);
            }
        }
    }
    __syncthreads();

    // ---------------- phase 6: window-reverse store (thread = channel) ----
    {
        const float* outFT = bufS;
        const int c = tid;
        const float* src = outFT + c * XSTR;
        float* base = out + (((size_t)b * CDIM + c) * HDIM + h0) * WDIM + w0;
#pragma unroll
        for (int r = 0; r < WIN; ++r) {
            float4 a0 = ((const float4*)(src + r * 8))[0];
            float4 a1 = ((const float4*)(src + r * 8))[1];
            float4* d4 = (float4*)(base + (size_t)r * WDIM);
            d4[0] = a0;
            d4[1] = a1;
        }
    }
}

// ---------------------------------------------------------------------------
extern "C" void kernel_launch(void* const* d_in, const int* in_sizes, int n_in,
                              void* d_out, int out_size, void* d_ws, size_t ws_size,
                              hipStream_t stream) {
    const float* x        = (const float*)d_in[0];
    const float* ln_gamma = (const float*)d_in[1];
    const float* ln_beta  = (const float*)d_in[2];
    const float* w_qkv    = (const float*)d_in[3];
    const float* b_qkv    = (const float*)d_in[4];
    const float* w_out    = (const float*)d_in[5];
    const float* b_out    = (const float*)d_in[6];
    float* out = (float*)d_out;

    _Float16* wbuf = (_Float16*)d_ws;   // f16 wqkv [0,196608), f16 wout [196608,262144)

    (void)hipFuncSetAttribute((const void*)win_attn_kernel,
                              hipFuncAttributeMaxDynamicSharedMemorySize,
                              SMEM_BYTES);

    cvt_weights_kernel<<<(WQKV_N + WOUT_N) / 256, 256, 0, stream>>>(w_qkv, w_out, wbuf);

    win_attn_kernel<<<NWIN_TOTAL, 256, SMEM_BYTES, stream>>>(
        x, ln_gamma, ln_beta, b_qkv, b_out,
        wbuf, wbuf + WQKV_N, out);
}
